// _MPNNet_50586124812376
// MI455X (gfx1250) — compile-verified, matrix operand order checked by reading, not run on hardware
//
#include <hip/hip_runtime.h>
#include <hip/hip_bf16.h>
#include <math.h>

#define NN 20000
#define NE 50000
#define NG 256
#define DIM 64
#define CONV_STEPS 12
#define S2S_STEPS 3

typedef __attribute__((ext_vector_type(16))) __bf16 v16bf;
typedef __attribute__((ext_vector_type(8)))  __bf16 v8bf;
typedef __attribute__((ext_vector_type(8)))  float  v8f;

__device__ __forceinline__ float leakyf(float v) { return v > 0.0f ? v : 0.01f * v; }
__device__ __forceinline__ float sigmf(float v)  { return 1.0f / (1.0f + __expf(-v)); }

// ---------------------------------------------------------------------------
// WMMA bf16 GEMM:  C[M,N] = act(A[M,K] @ Bt[N,K]^T + bias)
// A, Bt row-major bf16; M,N multiples of 16; K multiple of 32.
// One 16x16 output tile per wave32.
// ---------------------------------------------------------------------------
__device__ __forceinline__ v16bf frag_load(const __bf16* p) {
  v8bf lo = *reinterpret_cast<const v8bf*>(p);
  v8bf hi = *reinterpret_cast<const v8bf*>(p + 16);
  v16bf r;
#pragma unroll
  for (int i = 0; i < 8; ++i) { r[i] = lo[i]; r[i + 8] = hi[i]; }
  return r;
}

template <int ACT, int OUT_BF16>
__global__ __launch_bounds__(256)
void k_wmma_gemm(const __bf16* __restrict__ A, const __bf16* __restrict__ Bt,
                 const float* __restrict__ bias, void* __restrict__ Cout,
                 int M, int N, int K) {
  const int lane = threadIdx.x & 31;
  const int wave = (blockIdx.x * blockDim.x + threadIdx.x) >> 5;
  const int ntiles = N >> 4;
  const int mt = wave / ntiles;
  const int nt = wave - mt * ntiles;
  if (mt * 16 >= M) return;                      // whole-wave uniform exit
  const int kb = (lane >> 4) ? 8 : 0;            // K-half per lane group
  const __bf16* ap = A  + (size_t)(mt * 16 + (lane & 15)) * K + kb;
  const __bf16* bp = Bt + (size_t)(nt * 16 + (lane & 15)) * K + kb;
  v8f acc = {};
  for (int k = 0; k < K; k += 32) {
    v16bf af = frag_load(ap + k);
    v16bf bf = frag_load(bp + k);
    acc = __builtin_amdgcn_wmma_f32_16x16x32_bf16(false, af, false, bf,
                                                  (short)0, acc, false, false);
  }
  const int orow = mt * 16 + (lane >> 4) * 8;
  const int ocol = nt * 16 + (lane & 15);
  const float bv = bias ? bias[ocol] : 0.0f;
#pragma unroll
  for (int j = 0; j < 8; ++j) {
    float v = acc[j] + bv;
    if (ACT == 1) v = leakyf(v);
    const size_t idx = (size_t)(orow + j) * N + ocol;
    if (OUT_BF16) reinterpret_cast<__bf16*>(Cout)[idx] = (__bf16)v;
    else          reinterpret_cast<float*>(Cout)[idx]  = v;
  }
}

// ---------------------------------------------------------------------------
// Small helper kernels
// ---------------------------------------------------------------------------
__global__ void k_fill(float* __restrict__ p, int n, float v) {
  int t = blockIdx.x * blockDim.x + threadIdx.x;
  if (t < n) p[t] = v;
}

__global__ void k_f32_to_bf16(__bf16* __restrict__ d, const float* __restrict__ s, int n) {
  int t = blockIdx.x * blockDim.x + threadIdx.x;
  if (t < n) d[t] = (__bf16)s[t];
}

// pad rows from c_in to c_out with zeros, convert to bf16
__global__ void k_pad_rows(__bf16* __restrict__ d, const float* __restrict__ s,
                           int rows, int c_in, int c_out) {
  int t = blockIdx.x * blockDim.x + threadIdx.x;
  if (t >= rows * c_out) return;
  int r = t / c_out, c = t - r * c_out;
  d[t] = (c < c_in) ? (__bf16)s[r * c_in + c] : (__bf16)0.0f;
}

// dst[c*rows + r] = src[r*cols + c]   (produces Bt for out @ root_w)
__global__ void k_transpose_bf16(__bf16* __restrict__ d, const float* __restrict__ s,
                                 int rows, int cols) {
  int t = blockIdx.x * blockDim.x + threadIdx.x;
  if (t >= rows * cols) return;
  int r = t / cols, c = t - r * cols;
  d[c * rows + r] = (__bf16)s[t];
}

// edge net layer 1: H[e,j] = leaky(edge_attr[e,:4] @ w1[j,:4] + b1[j]) -> bf16
__global__ void k_edge_net1(__bf16* __restrict__ H, const float* __restrict__ ea,
                            const float* __restrict__ w1, const float* __restrict__ b1) {
  int t = blockIdx.x * blockDim.x + threadIdx.x;
  if (t >= NE * 128) return;
  int e = t >> 7, j = t & 127;
  float acc = b1[j];
#pragma unroll
  for (int k = 0; k < 4; ++k) acc += ea[e * 4 + k] * w1[j * 4 + k];
  H[t] = (__bf16)leakyf(acc);
}

__global__ void k_deg_count(float* __restrict__ deg, const int* __restrict__ ei) {
  int e = blockIdx.x * blockDim.x + threadIdx.x;
  if (e < NE) atomicAdd(&deg[ei[NE + e]], 1.0f);
}

__global__ void k_deg_inv(float* __restrict__ rdeg) {
  int t = blockIdx.x * blockDim.x + threadIdx.x;
  if (t >= NN) return;
  float d = rdeg[t];
  rdeg[t] = 1.0f / (d < 1.0f ? 1.0f : d);
}

// msg[e,o] = sum_i feat[src[e],i] * w_e[e,i,o]; atomic scatter-add into agg[dst]
// 4 edges / 256-thread block, 64 threads per edge. Coalesced bf16 reads of w_e.
__global__ __launch_bounds__(256)
void k_msg(float* __restrict__ agg, const float* __restrict__ feat,
           const __bf16* __restrict__ we, const int* __restrict__ ei) {
  __shared__ float xs[4][DIM];
  const int sub = threadIdx.x >> 6;
  const int t   = threadIdx.x & 63;
  const int e   = blockIdx.x * 4 + sub;
  const int src = ei[e];
  const int dst = ei[NE + e];
  xs[sub][t] = feat[src * DIM + t];
  __syncthreads();
  const __bf16* w = we + (size_t)e * (DIM * DIM);
  __builtin_prefetch(w + DIM * DIM + t * DIM, 0, 1);  // next edge's matrix
  float acc = 0.0f;
#pragma unroll 8
  for (int i = 0; i < DIM; ++i) acc += xs[sub][i] * (float)w[i * DIM + t];
  atomicAdd(&agg[dst * DIM + t], acc);
}

// m = leaky(agg/deg + out@root_w + conv_b)  -> bf16 (GEMM input)
__global__ void k_node1(__bf16* __restrict__ mb, const float* __restrict__ agg,
                        const float* __restrict__ troot, const float* __restrict__ rdeg,
                        const float* __restrict__ conv_b) {
  int t = blockIdx.x * blockDim.x + threadIdx.x;
  if (t >= NN * DIM) return;
  int n = t >> 6, d = t & 63;
  mb[t] = (__bf16)leakyf(agg[t] * rdeg[n] + troot[t] + conv_b[d]);
}

// GRU cell elementwise; updates feat (=h=out) and its bf16 mirror
__global__ void k_gru(float* __restrict__ feat, __bf16* __restrict__ featb,
                      const float* __restrict__ gi, const float* __restrict__ gh) {
  int t = blockIdx.x * blockDim.x + threadIdx.x;
  if (t >= NN * DIM) return;
  int n = t >> 6, d = t & 63;
  int base = n * 192;
  float ir = gi[base + d], iz = gi[base + 64 + d], inn = gi[base + 128 + d];
  float hr = gh[base + d], hz = gh[base + 64 + d], hnn = gh[base + 128 + d];
  float r = sigmf(ir + hr);
  float z = sigmf(iz + hz);
  float nv = tanhf(inn + r * hnn);
  float hv = (1.0f - z) * nv + z * feat[t];
  feat[t]  = hv;
  featb[t] = (__bf16)hv;
}

// ---------------------------------------------------------------------------
// Set2Set (256 graphs — tiny)
// ---------------------------------------------------------------------------
__global__ __launch_bounds__(256)
void k_lstm_gates(float* __restrict__ gates, const float* __restrict__ qstar,
                  const float* __restrict__ hl, const float* __restrict__ wih,
                  const float* __restrict__ whh, const float* __restrict__ bih,
                  const float* __restrict__ bhh) {
  int g = blockIdx.x, j = threadIdx.x;     // 256 x 256
  float acc = bih[j] + bhh[j];
  for (int k = 0; k < 128; ++k) acc += wih[j * 128 + k] * qstar[g * 128 + k];
  for (int k = 0; k < 64; ++k)  acc += whh[j * 64 + k]  * hl[g * 64 + k];
  gates[g * 256 + j] = acc;
}

__global__ void k_lstm_update(float* __restrict__ hl, float* __restrict__ cl,
                              const float* __restrict__ gates) {
  int t = blockIdx.x * blockDim.x + threadIdx.x;
  if (t >= NG * DIM) return;
  int g = t >> 6, d = t & 63;
  const float* gt = gates + g * 256;
  float i = sigmf(gt[d]), f = sigmf(gt[64 + d]);
  float gg = tanhf(gt[128 + d]), o = sigmf(gt[192 + d]);
  float c = f * cl[t] + i * gg;
  cl[t] = c;
  hl[t] = o * tanhf(c);
}

__device__ __forceinline__ void atomicMaxF(float* a, float v) {
  if (v >= 0.0f) atomicMax((int*)a, __float_as_int(v));
  else           atomicMin((unsigned int*)a, __float_as_uint(v));
}

__global__ void k_attn_e(float* __restrict__ evec, float* __restrict__ emax,
                         const float* __restrict__ feat, const float* __restrict__ hl,
                         const int* __restrict__ batch) {
  int n = blockIdx.x * blockDim.x + threadIdx.x;
  if (n >= NN) return;
  int b = batch[n];
  float acc = 0.0f;
#pragma unroll 8
  for (int d = 0; d < DIM; ++d) acc += feat[n * DIM + d] * hl[b * DIM + d];
  evec[n] = acc;
  atomicMaxF(&emax[b], acc);
}

__global__ void k_attn_a(float* __restrict__ avec, float* __restrict__ asum,
                         const float* __restrict__ evec, const float* __restrict__ emax,
                         const int* __restrict__ batch) {
  int n = blockIdx.x * blockDim.x + threadIdx.x;
  if (n >= NN) return;
  int b = batch[n];
  float av = __expf(evec[n] - emax[b]);
  avec[n] = av;
  atomicAdd(&asum[b], av);
}

__global__ void k_attn_r(float* __restrict__ rread, const float* __restrict__ avec,
                         const float* __restrict__ asum, const float* __restrict__ feat,
                         const int* __restrict__ batch) {
  int t = blockIdx.x * blockDim.x + threadIdx.x;
  if (t >= NN * DIM) return;
  int n = t >> 6, d = t & 63;
  int b = batch[n];
  atomicAdd(&rread[b * DIM + d], avec[n] / (asum[b] + 1e-16f) * feat[t]);
}

__global__ void k_qstar(float* __restrict__ qstar, const float* __restrict__ hl,
                        const float* __restrict__ rread) {
  int t = blockIdx.x * blockDim.x + threadIdx.x;
  if (t >= NG * 128) return;
  int g = t >> 7, j = t & 127;
  qstar[t] = (j < 64) ? hl[g * 64 + j] : rread[g * 64 + (j - 64)];
}

__global__ void k_final(float* __restrict__ out, const float* __restrict__ qstar,
                        const float* __restrict__ w, const float* __restrict__ b) {
  int g = blockIdx.x * blockDim.x + threadIdx.x;
  if (g >= NG) return;
  float acc = b[0];
  for (int k = 0; k < 128; ++k) acc += w[k] * qstar[g * 128 + k];
  out[g] = acc;
}

// ---------------------------------------------------------------------------
// Host orchestration
// ---------------------------------------------------------------------------
static inline int G(int n) { return (n + 255) / 256; }

extern "C" void kernel_launch(void* const* d_in, const int* in_sizes, int n_in,
                              void* d_out, int out_size, void* d_ws, size_t ws_size,
                              hipStream_t stream) {
  const float* x         = (const float*)d_in[0];
  const float* edge_attr = (const float*)d_in[1];
  const int*   edge_idx  = (const int*)d_in[2];
  const int*   batch     = (const int*)d_in[3];
  const float* lin0_w = (const float*)d_in[4];
  const float* lin0_b = (const float*)d_in[5];
  const float* net_w1 = (const float*)d_in[6];
  const float* net_b1 = (const float*)d_in[7];
  const float* net_w2 = (const float*)d_in[8];
  const float* net_b2 = (const float*)d_in[9];
  const float* root_w = (const float*)d_in[10];
  const float* conv_b = (const float*)d_in[11];
  const float* gwih = (const float*)d_in[12];
  const float* gwhh = (const float*)d_in[13];
  const float* gbih = (const float*)d_in[14];
  const float* gbhh = (const float*)d_in[15];
  const float* lwih = (const float*)d_in[16];
  const float* lwhh = (const float*)d_in[17];
  const float* lbih = (const float*)d_in[18];
  const float* lbhh = (const float*)d_in[19];
  const float* lin3_w = (const float*)d_in[20];
  const float* lin3_b = (const float*)d_in[21];

  char* wsb = (char*)d_ws;
  size_t off = 0;
  auto alloc = [&](size_t bytes) -> void* {
    void* p = wsb + off;
    off = (off + bytes + 255) & ~(size_t)255;
    return p;
  };

  __bf16* web    = (__bf16*)alloc((size_t)NE * 4096 * 2);  // w_e bf16 (410 MB)
  __bf16* Hb     = (__bf16*)alloc((size_t)NE * 128 * 2);
  __bf16* xpad   = (__bf16*)alloc((size_t)NN * 96 * 2);
  __bf16* lin0wp = (__bf16*)alloc((size_t)64 * 96 * 2);
  __bf16* w2b    = (__bf16*)alloc((size_t)4096 * 128 * 2);
  __bf16* rootTb = (__bf16*)alloc((size_t)64 * 64 * 2);
  __bf16* wihb   = (__bf16*)alloc((size_t)192 * 64 * 2);
  __bf16* whhb   = (__bf16*)alloc((size_t)192 * 64 * 2);
  float*  feat   = (float*)alloc((size_t)NN * DIM * 4);    // out == h
  __bf16* featb  = (__bf16*)alloc((size_t)NN * DIM * 2);
  __bf16* mb     = (__bf16*)alloc((size_t)NN * DIM * 2);
  float*  agg    = (float*)alloc((size_t)NN * DIM * 4);
  float*  troot  = (float*)alloc((size_t)NN * DIM * 4);
  float*  gi     = (float*)alloc((size_t)NN * 192 * 4);
  float*  gh     = (float*)alloc((size_t)NN * 192 * 4);
  float*  rdeg   = (float*)alloc((size_t)NN * 4);
  float*  evec   = (float*)alloc((size_t)NN * 4);
  float*  avec   = (float*)alloc((size_t)NN * 4);
  float*  emax   = (float*)alloc((size_t)NG * 4);
  float*  asum   = (float*)alloc((size_t)NG * 4);
  float*  rread  = (float*)alloc((size_t)NG * DIM * 4);
  float*  qstar  = (float*)alloc((size_t)NG * 128 * 4);
  float*  hl     = (float*)alloc((size_t)NG * DIM * 4);
  float*  cl     = (float*)alloc((size_t)NG * DIM * 4);
  float*  gates  = (float*)alloc((size_t)NG * 256 * 4);

  // ---- weight prep (bf16) ----
  k_f32_to_bf16<<<G(4096 * 128), 256, 0, stream>>>(w2b, net_w2, 4096 * 128);
  k_f32_to_bf16<<<G(192 * 64), 256, 0, stream>>>(wihb, gwih, 192 * 64);
  k_f32_to_bf16<<<G(192 * 64), 256, 0, stream>>>(whhb, gwhh, 192 * 64);
  k_transpose_bf16<<<G(64 * 64), 256, 0, stream>>>(rootTb, root_w, 64, 64);
  k_pad_rows<<<G(NN * 96), 256, 0, stream>>>(xpad, x, NN, 72, 96);
  k_pad_rows<<<G(64 * 96), 256, 0, stream>>>(lin0wp, lin0_w, 64, 72, 96);

  // ---- lin0: feat = leaky(x @ lin0_w.T + b)   [WMMA]  M=20000 N=64 K=96 ----
  k_wmma_gemm<1, 0><<<(1250 * 4) / 8, 256, 0, stream>>>(xpad, lin0wp, lin0_b, feat,
                                                        NN, 64, 96);
  k_f32_to_bf16<<<G(NN * DIM), 256, 0, stream>>>(featb, feat, NN * DIM);

  // ---- edge network: w_e bf16  [WMMA]  M=50000 N=4096 K=128 ----
  k_edge_net1<<<G(NE * 128), 256, 0, stream>>>(Hb, edge_attr, net_w1, net_b1);
  k_wmma_gemm<0, 1><<<(3125 * 256) / 8, 256, 0, stream>>>(Hb, w2b, net_b2, web,
                                                          NE, 4096, 128);

  // ---- degree ----
  k_fill<<<G(NN), 256, 0, stream>>>(rdeg, NN, 0.0f);
  k_deg_count<<<G(NE), 256, 0, stream>>>(rdeg, edge_idx);
  k_deg_inv<<<G(NN), 256, 0, stream>>>(rdeg);

  // ---- 12 conv + GRU steps ----
  for (int s = 0; s < CONV_STEPS; ++s) {
    k_fill<<<G(NN * DIM), 256, 0, stream>>>(agg, NN * DIM, 0.0f);
    k_msg<<<NE / 4, 256, 0, stream>>>(agg, feat, web, edge_idx);
    k_wmma_gemm<0, 0><<<(1250 * 4) / 8, 256, 0, stream>>>(featb, rootTb, nullptr,
                                                          troot, NN, 64, 64);
    k_node1<<<G(NN * DIM), 256, 0, stream>>>(mb, agg, troot, rdeg, conv_b);
    k_wmma_gemm<0, 0><<<(1250 * 12) / 8, 256, 0, stream>>>(mb, wihb, gbih, gi,
                                                           NN, 192, 64);
    k_wmma_gemm<0, 0><<<(1250 * 12) / 8, 256, 0, stream>>>(featb, whhb, gbhh, gh,
                                                           NN, 192, 64);
    k_gru<<<G(NN * DIM), 256, 0, stream>>>(feat, featb, gi, gh);
  }

  // ---- Set2Set ----
  k_fill<<<G(NG * 128), 256, 0, stream>>>(qstar, NG * 128, 0.0f);
  k_fill<<<G(NG * DIM), 256, 0, stream>>>(hl, NG * DIM, 0.0f);
  k_fill<<<G(NG * DIM), 256, 0, stream>>>(cl, NG * DIM, 0.0f);
  for (int s = 0; s < S2S_STEPS; ++s) {
    k_lstm_gates<<<NG, 256, 0, stream>>>(gates, qstar, hl, lwih, lwhh, lbih, lbhh);
    k_lstm_update<<<G(NG * DIM), 256, 0, stream>>>(hl, cl, gates);
    k_fill<<<G(NG), 256, 0, stream>>>(emax, NG, -3.4e38f);
    k_fill<<<G(NG), 256, 0, stream>>>(asum, NG, 0.0f);
    k_fill<<<G(NG * DIM), 256, 0, stream>>>(rread, NG * DIM, 0.0f);
    k_attn_e<<<G(NN), 256, 0, stream>>>(evec, emax, feat, hl, batch);
    k_attn_a<<<G(NN), 256, 0, stream>>>(avec, asum, evec, emax, batch);
    k_attn_r<<<G(NN * DIM), 256, 0, stream>>>(rread, avec, asum, feat, batch);
    k_qstar<<<G(NG * 128), 256, 0, stream>>>(qstar, hl, rread);
  }
  k_final<<<G(NG), 256, 0, stream>>>((float*)d_out, qstar, lin3_w, lin3_b);
}
